// MultiHeadAttention_46943992545524
// MI455X (gfx1250) — matrix operand order repaired and checked by reading, not compiled, not run on hardware
//
#include <hip/hip_runtime.h>
#include <hip/hip_bf16.h>

typedef __bf16 bf16_t;
typedef __attribute__((ext_vector_type(16))) __bf16 v16bf;
typedef __attribute__((ext_vector_type(8)))  float   v8f;
typedef __attribute__((ext_vector_type(2)))  __bf16 bf16x2;

#define B_   2
#define S_   2048
#define D_   1024
#define NH_  16
#define KV_  4
#define HD_  64
#define G_   (NH_ / KV_)      // 4
#define GKV_ (KV_ * HD_)      // 256

// ---------------------------------------------------------------- converts
__global__ void cvt_f32_bf16(const float* __restrict__ src,
                             bf16_t* __restrict__ dst, int n) {
  int i = blockIdx.x * blockDim.x + threadIdx.x;
  if (i < n) dst[i] = (bf16_t)src[i];
}

// W [K,N] f32 -> Wt [N,K] bf16
__global__ void transpose_f32_bf16(const float* __restrict__ W,
                                   bf16_t* __restrict__ Wt, int K, int N) {
  int i = blockIdx.x * blockDim.x + threadIdx.x;
  if (i < K * N) {
    int k = i / N, n = i % N;
    Wt[(size_t)n * K + k] = (bf16_t)W[i];
  }
}

// ---------------------------------------------------------------- GEMM
// C[M,N] = A[M,K] @ Bt[N,K]^T + bias
// one wave computes 64x32 (4 M-tiles x 2 N-tiles); K loop ping-pong unrolled
// by 64 so double-buffering needs no register copies. Requires K % 64 == 0.
__global__ __launch_bounds__(128) void gemm_bf16_wmma(
    const bf16_t* __restrict__ A, const bf16_t* __restrict__ Bt,
    const float* __restrict__ bias, float* __restrict__ C,
    int M, int N, int K) {
  const int lane = threadIdx.x & 31;
  const int wave = threadIdx.x >> 5;
  const int li   = lane & 15;
  const int grp  = lane >> 4;
  const int m0 = (blockIdx.y * 4 + wave) * 64;
  const int n0 = blockIdx.x * 32;

  const bf16_t* arow[4];
#pragma unroll
  for (int mt = 0; mt < 4; ++mt)
    arow[mt] = A + (size_t)(m0 + mt * 16 + li) * K + grp * 16;
  const bf16_t* brow[2];
#pragma unroll
  for (int nt = 0; nt < 2; ++nt)
    brow[nt] = Bt + (size_t)(n0 + nt * 16 + li) * K + grp * 16;

  auto loadA = [&](int mt, int k0) {
    v16bf a;
#pragma unroll
    for (int i = 0; i < 8; ++i) {                 // A: K = 16*grp + e
      int ka = 2 * i;
      bf16x2 p = *(const bf16x2*)(arow[mt] + k0 + ka);
      a[2 * i] = p.x; a[2 * i + 1] = p.y;
    }
    return a;
  };
  auto loadB = [&](int nt, int k0) {
    v16bf b;
#pragma unroll
    for (int i = 0; i < 8; ++i) {                 // B: K = 16*grp + e
      bf16x2 p = *(const bf16x2*)(brow[nt] + k0 + 2 * i);
      b[2 * i] = p.x; b[2 * i + 1] = p.y;
    }
    return b;
  };

  v8f acc[4][2] = {};
  auto mma8 = [&](v16bf (&a)[4], v16bf (&b)[2]) {
#pragma unroll
    for (int mt = 0; mt < 4; ++mt)
#pragma unroll
      for (int nt = 0; nt < 2; ++nt)
        acc[mt][nt] = __builtin_amdgcn_wmma_f32_16x16x32_bf16(
            false, a[mt], false, b[nt], (short)0, acc[mt][nt], false, false);
  };

  v16bf aA[4], bA[2], aB[4], bB[2];
#pragma unroll
  for (int mt = 0; mt < 4; ++mt) aA[mt] = loadA(mt, 0);
#pragma unroll
  for (int nt = 0; nt < 2; ++nt) bA[nt] = loadB(nt, 0);

  int k0 = 0;
  for (; k0 < K - 64; k0 += 64) {
    __builtin_prefetch(arow[0] + k0 + 512, 0, 0);  // global_prefetch_b8
#pragma unroll
    for (int mt = 0; mt < 4; ++mt) aB[mt] = loadA(mt, k0 + 32);
#pragma unroll
    for (int nt = 0; nt < 2; ++nt) bB[nt] = loadB(nt, k0 + 32);
    mma8(aA, bA);
#pragma unroll
    for (int mt = 0; mt < 4; ++mt) aA[mt] = loadA(mt, k0 + 64);
#pragma unroll
    for (int nt = 0; nt < 2; ++nt) bA[nt] = loadB(nt, k0 + 64);
    mma8(aB, bB);
  }
  // tail: the last two K-steps (K % 64 == 0)
#pragma unroll
  for (int mt = 0; mt < 4; ++mt) aB[mt] = loadA(mt, k0 + 32);
#pragma unroll
  for (int nt = 0; nt < 2; ++nt) bB[nt] = loadB(nt, k0 + 32);
  mma8(aA, bA);
  mma8(aB, bB);

  float bv0 = bias[n0 + li];
  float bv1 = bias[n0 + 16 + li];
#pragma unroll
  for (int mt = 0; mt < 4; ++mt)
#pragma unroll
    for (int r = 0; r < 8; ++r) {
      float* cr = C + (size_t)(m0 + mt * 16 + grp * 8 + r) * N + n0 + li;
      cr[0]  = acc[mt][0][r] + bv0;
      cr[16] = acc[mt][1][r] + bv1;
    }
}

// ---------------------------------------------------------------- RoPE + pack
// lin [B,S,H*HD] f32 -> out [B,H,S,HD] bf16, one thread per rotation pair
__global__ void rope_pack(const float* __restrict__ lin,
                          bf16_t* __restrict__ out, int H) {
  int t = blockIdx.x * blockDim.x + threadIdx.x;
  int j = t & 31; t >>= 5;
  int h = t % H;  t /= H;
  int s = t % S_;
  int b = t / S_;
  float inv = __powf(5000.0f, -(float)j * (1.0f / 32.0f));
  float ang = (float)s * inv;
  float sn, cs;
  __sincosf(ang, &sn, &cs);
  const float* p = lin + (((size_t)(b * S_ + s)) * H + h) * HD_;
  float t1 = p[2 * j], t2 = p[2 * j + 1];
  bf16_t* q = out + (((size_t)(b * H + h)) * S_ + s) * HD_;
  q[2 * j]     = (bf16_t)(t1 * cs - t2 * sn);
  q[2 * j + 1] = (bf16_t)(t1 * sn + t2 * cs);
}

// vlin [B,S,KV*HD] f32 -> vt [B,KV,HD,S] bf16 (transposed for P@V B-fragments)
__global__ void v_pack(const float* __restrict__ lin,
                       bf16_t* __restrict__ vt) {
  int t = blockIdx.x * blockDim.x + threadIdx.x;
  int d = t % HD_; int r = t / HD_;
  int kv = r % KV_; r /= KV_;
  int s = r % S_;  int b = r / S_;
  vt[(((size_t)(b * KV_ + kv)) * HD_ + d) * S_ + s] = (bf16_t)lin[t];
}

// ---------------------------------------------------------------- attention
// q [B,NH,S,HD], k [B,KV,S,HD], vt [B,KV,HD,S] -> ctx [B,S,NH*HD] bf16
// one wave = 32 queries of one head (2 q-tiles sharing K/V fragments),
// flash loop over S in tiles of 32 keys
__global__ __launch_bounds__(128) void attn_wmma(
    const bf16_t* __restrict__ q, const bf16_t* __restrict__ k,
    const bf16_t* __restrict__ vt, bf16_t* __restrict__ ctx) {
  __shared__ __align__(16) bf16_t lds[4][2][16 * 32];
  const int lane = threadIdx.x & 31;
  const int wave = threadIdx.x >> 5;
  const int li = lane & 15, grp = lane >> 4;
  const int h  = blockIdx.y;
  const int b  = blockIdx.z;
  const int kv = h / G_;
  const int q0 = (blockIdx.x * 4 + wave) * 32;

  // Q A-fragments: [q-pair][head-dim half]
  v16bf aq[2][2];
#pragma unroll
  for (int qp = 0; qp < 2; ++qp) {
    const bf16_t* qrow =
        q + (((size_t)(b * NH_ + h)) * S_ + q0 + qp * 16 + li) * HD_ + grp * 16;
#pragma unroll
    for (int i = 0; i < 8; ++i) {
      int ka = 2 * i;
      bf16x2 p0 = *(const bf16x2*)(qrow + ka);
      bf16x2 p1 = *(const bf16x2*)(qrow + 32 + ka);
      aq[qp][0][2 * i] = p0.x; aq[qp][0][2 * i + 1] = p0.y;
      aq[qp][1][2 * i] = p1.x; aq[qp][1][2 * i + 1] = p1.y;
    }
  }

  float mrow[2][8], lrow[2][8];
#pragma unroll
  for (int qp = 0; qp < 2; ++qp)
#pragma unroll
    for (int r = 0; r < 8; ++r) { mrow[qp][r] = -3.0e38f; lrow[qp][r] = 0.0f; }
  v8f o[2][4] = {{{}, {}, {}, {}}, {{}, {}, {}, {}}};
  const float scale = 0.125f;  // 1/sqrt(64)

  const bf16_t* kbase = k  + ((size_t)(b * KV_ + kv)) * S_ * HD_;
  const bf16_t* vbase = vt + ((size_t)(b * KV_ + kv)) * HD_ * S_;

  for (int s0 = 0; s0 < S_; s0 += 32) {
    // K B-fragments, shared by both q-tiles: [key nt][head-dim half]
    v16bf kb[2][2];
#pragma unroll
    for (int nt = 0; nt < 2; ++nt) {
      const bf16_t* krow =
          kbase + (size_t)(s0 + nt * 16 + li) * HD_ + grp * 16;
#pragma unroll
      for (int i = 0; i < 8; ++i) {
        bf16x2 p0 = *(const bf16x2*)(krow + 2 * i);
        bf16x2 p1 = *(const bf16x2*)(krow + 32 + 2 * i);
        kb[nt][0][2 * i] = p0.x; kb[nt][0][2 * i + 1] = p0.y;
        kb[nt][1][2 * i] = p1.x; kb[nt][1][2 * i + 1] = p1.y;
      }
    }
    // V B-fragments hoisted: issue early so they overlap the softmax VALU work
    v16bf vb[4];
#pragma unroll
    for (int nt = 0; nt < 4; ++nt) {
      const bf16_t* vrow = vbase + (size_t)(nt * 16 + li) * S_ + s0 + grp * 16;
#pragma unroll
      for (int i = 0; i < 8; ++i) {
        bf16x2 p = *(const bf16x2*)(vrow + 2 * i);
        vb[nt][2 * i] = p.x; vb[nt][2 * i + 1] = p.y;
      }
    }

#pragma unroll
    for (int qp = 0; qp < 2; ++qp) {
      v8f sf[2];
#pragma unroll
      for (int nt = 0; nt < 2; ++nt) {
        v8f sc = {};
        sc = __builtin_amdgcn_wmma_f32_16x16x32_bf16(
            false, aq[qp][0], false, kb[nt][0], (short)0, sc, false, false);
        sc = __builtin_amdgcn_wmma_f32_16x16x32_bf16(
            false, aq[qp][1], false, kb[nt][1], (short)0, sc, false, false);
        sf[nt] = sc;
      }

      // online softmax; row r lives in VGPR r across one 16-lane group
      bf16_t* myP = lds[wave][qp];
#pragma unroll
      for (int r = 0; r < 8; ++r) {
        float s0v = sf[0][r] * scale, s1v = sf[1][r] * scale;
        float mx = fmaxf(s0v, s1v);
        mx = fmaxf(mx, __shfl_xor(mx, 1));
        mx = fmaxf(mx, __shfl_xor(mx, 2));
        mx = fmaxf(mx, __shfl_xor(mx, 4));
        mx = fmaxf(mx, __shfl_xor(mx, 8));
        float mnew = fmaxf(mrow[qp][r], mx);
        float corr = __expf(mrow[qp][r] - mnew);
        float p0 = __expf(s0v - mnew);
        float p1 = __expf(s1v - mnew);
        float ps = p0 + p1;
        ps += __shfl_xor(ps, 1);
        ps += __shfl_xor(ps, 2);
        ps += __shfl_xor(ps, 4);
        ps += __shfl_xor(ps, 8);
        lrow[qp][r] = lrow[qp][r] * corr + ps;
        mrow[qp][r] = mnew;
#pragma unroll
        for (int nt = 0; nt < 4; ++nt) o[qp][nt][r] *= corr;
        myP[(r + grp * 8) * 32 + li]      = (bf16_t)p0;  // D-layout -> row major
        myP[(r + grp * 8) * 32 + 16 + li] = (bf16_t)p1;
      }
    }

    // re-load P as A-fragments (per-wave LDS region; LDS ops in-order per wave)
    v16bf ap[2];
#pragma unroll
    for (int qp = 0; qp < 2; ++qp) {
      const bf16_t* prow = lds[wave][qp] + li * 32 + grp * 16;
#pragma unroll
      for (int i = 0; i < 8; ++i) {
        int ka = 2 * i;
        bf16x2 p = *(const bf16x2*)(prow + ka);
        ap[qp][2 * i] = p.x; ap[qp][2 * i + 1] = p.y;
      }
    }

    // O += P @ V over 4 head-dim tiles; V fragment shared by both q-tiles
#pragma unroll
    for (int nt = 0; nt < 4; ++nt) {
      o[0][nt] = __builtin_amdgcn_wmma_f32_16x16x32_bf16(
          false, ap[0], false, vb[nt], (short)0, o[0][nt], false, false);
      o[1][nt] = __builtin_amdgcn_wmma_f32_16x16x32_bf16(
          false, ap[1], false, vb[nt], (short)0, o[1][nt], false, false);
    }
  }

  // normalize and write context in [B,S,NH*HD] (ready as A for output GEMM)
#pragma unroll
  for (int qp = 0; qp < 2; ++qp)
#pragma unroll
    for (int r = 0; r < 8; ++r) {
      float inv = 1.0f / lrow[qp][r];
      int row = q0 + qp * 16 + grp * 8 + r;
      bf16_t* cr = ctx + ((size_t)(b * S_) + row) * (size_t)(NH_ * HD_) +
                   h * HD_ + li;
      cr[0]  = (bf16_t)(o[qp][0][r] * inv);
      cr[16] = (bf16_t)(o[qp][1][r] * inv);
      cr[32] = (bf16_t)(o[qp][2][r] * inv);
      cr[48] = (bf16_t)(o[qp][3][r] * inv);
    }
}

// ---------------------------------------------------------------- launcher
extern "C" void kernel_launch(void* const* d_in, const int* in_sizes, int n_in,
                              void* d_out, int out_size, void* d_ws,
                              size_t ws_size, hipStream_t stream) {
  const float* x  = (const float*)d_in[0];
  const float* y  = (const float*)d_in[1];
  const float* Wq = (const float*)d_in[2];
  const float* bq = (const float*)d_in[3];
  const float* Wk = (const float*)d_in[4];
  const float* bk = (const float*)d_in[5];
  const float* Wv = (const float*)d_in[6];
  const float* bv = (const float*)d_in[7];
  const float* Wo = (const float*)d_in[8];
  const float* bo = (const float*)d_in[9];
  float* out = (float*)d_out;

  char* ws = (char*)d_ws;
  size_t off = 0;
  auto alloc = [&](size_t bytes) {
    void* p = ws + off;
    off += (bytes + 255) & ~(size_t)255;
    return p;
  };
  const size_t MS = (size_t)B_ * S_;  // 4096
  bf16_t* xb   = (bf16_t*)alloc(MS * D_ * 2);
  bf16_t* yb   = (bf16_t*)alloc(MS * D_ * 2);
  bf16_t* Wqt  = (bf16_t*)alloc((size_t)D_ * D_ * 2);
  bf16_t* Wkt  = (bf16_t*)alloc((size_t)D_ * GKV_ * 2);
  bf16_t* Wvt  = (bf16_t*)alloc((size_t)D_ * GKV_ * 2);
  bf16_t* Wot  = (bf16_t*)alloc((size_t)D_ * D_ * 2);
  float*  qlin = (float*)alloc(MS * D_ * 4);
  float*  klin = (float*)alloc(MS * GKV_ * 4);
  float*  vlin = (float*)alloc(MS * GKV_ * 4);
  bf16_t* qr   = (bf16_t*)alloc(MS * D_ * 2);
  bf16_t* kr   = (bf16_t*)alloc(MS * GKV_ * 2);
  bf16_t* vtr  = (bf16_t*)alloc(MS * GKV_ * 2);
  bf16_t* ctx  = (bf16_t*)alloc(MS * D_ * 2);

  const int nx = B_ * S_ * D_;
  cvt_f32_bf16<<<nx / 256, 256, 0, stream>>>(x, xb, nx);
  cvt_f32_bf16<<<nx / 256, 256, 0, stream>>>(y, yb, nx);
  transpose_f32_bf16<<<(D_ * D_) / 256, 256, 0, stream>>>(Wq, Wqt, D_, D_);
  transpose_f32_bf16<<<(D_ * GKV_) / 256, 256, 0, stream>>>(Wk, Wkt, D_, GKV_);
  transpose_f32_bf16<<<(D_ * GKV_) / 256, 256, 0, stream>>>(Wv, Wvt, D_, GKV_);
  transpose_f32_bf16<<<(D_ * D_) / 256, 256, 0, stream>>>(Wo, Wot, D_, D_);

  dim3 blk(128);
  // 4 waves/block, each wave 64x32 -> 256 rows x 32 cols per block
  gemm_bf16_wmma<<<dim3(D_ / 32, MS / 256), blk, 0, stream>>>(
      xb, Wqt, bq, qlin, (int)MS, D_, D_);
  gemm_bf16_wmma<<<dim3(GKV_ / 32, MS / 256), blk, 0, stream>>>(
      yb, Wkt, bk, klin, (int)MS, GKV_, D_);
  gemm_bf16_wmma<<<dim3(GKV_ / 32, MS / 256), blk, 0, stream>>>(
      yb, Wvt, bv, vlin, (int)MS, GKV_, D_);

  rope_pack<<<(B_ * S_ * NH_ * 32) / 256, 256, 0, stream>>>(qlin, qr, NH_);
  rope_pack<<<(B_ * S_ * KV_ * 32) / 256, 256, 0, stream>>>(klin, kr, KV_);
  v_pack<<<(B_ * S_ * GKV_) / 256, 256, 0, stream>>>(vlin, vtr);

  // 4 waves/block, each wave 32 queries -> 128 queries per block
  attn_wmma<<<dim3(S_ / 128, NH_, B_), blk, 0, stream>>>(qr, kr, vtr, ctx);

  gemm_bf16_wmma<<<dim3(D_ / 32, MS / 256), blk, 0, stream>>>(
      ctx, Wot, bo, out, (int)MS, D_, D_);
}
